// MHAAttentionBlock_17317308137775
// MI455X (gfx1250) — matrix operand order repaired and checked by reading, not compiled, not run on hardware
//
#include <hip/hip_runtime.h>

typedef __attribute__((ext_vector_type(16))) __bf16 v16bf;
typedef __attribute__((ext_vector_type(8)))  float  v8f;
typedef unsigned short u16;
typedef unsigned int   u32;
typedef unsigned long long u64;

#define BATCH 4
#define SEQ   4096
#define CH    256
#define NTOK  (BATCH * SEQ)      // 16384
#define KPAD  264                // K chunk LDS row stride (u16), bank-safe
#define VPAD  40                 // V chunk LDS row stride (u16), bank-safe

union ABOp { uint4 u[2]; v16bf v; };

__device__ __forceinline__ u16 f2bf(float f) {
  u32 u = __float_as_uint(f);
  u32 r = (u + 0x7FFFu + ((u >> 16) & 1u)) >> 16;   // RNE
  return (u16)r;
}

// A operand 16x32 bf16 from row-major [row][k], k contiguous.
__device__ __forceinline__ v16bf load_a16(const u16* base, int stride, int row0, int k0, int lane) {
  const int half = lane >> 4, r = lane & 15;
  const u16* p = base + (size_t)(row0 + r) * stride + k0 + half * 8;
  ABOp op;
  op.u[0] = *reinterpret_cast<const uint4*>(p);
  op.u[1] = *reinterpret_cast<const uint4*>(p + 16);
  return op.v;
}

// B operand 32x16 bf16 from K-contiguous storage [col][k].
__device__ __forceinline__ v16bf load_b16(const u16* base, int stride, int col0, int k0, int lane) {
  const int half = lane >> 4, c = lane & 15;
  const u16* p = base + (size_t)(col0 + c) * stride + k0 + half * 8;
  ABOp op;
  op.u[0] = *reinterpret_cast<const uint4*>(p);
  op.u[1] = *reinterpret_cast<const uint4*>(p + 16);
  return op.v;
}

__device__ __forceinline__ v8f wmma_bf(v16bf a, v16bf b, v8f c) {
  return __builtin_amdgcn_wmma_f32_16x16x32_bf16(false, a, false, b, (short)0, c, false, false);
}

#if __has_builtin(__builtin_amdgcn_s_wait_asynccnt)
#define WAIT_ASYNC(n) __builtin_amdgcn_s_wait_asynccnt(n)
#else
#define WAIT_ASYNC(n) asm volatile("s_wait_asynccnt %0" :: "i"(n) : "memory")
#endif

// gfx1250 async copy: global -> LDS, 16 bytes per lane, tracked by ASYNCcnt
__device__ __forceinline__ void async_ld16(const u16* gp, u16* lp) {
  u32 la = (u32)(size_t)lp;                 // low 32 bits of flat LDS addr = wave-relative LDS offset
  u64 ga = (u64)(size_t)gp;
  asm volatile("global_load_async_to_lds_b128 %0, %1, off" :: "v"(la), "v"(ga) : "memory");
}

// ---------------- weight prep: transpose + bf16 convert ----------------
__global__ void wprep_kernel(const float* __restrict__ wqkv, const float* __restrict__ wout,
                             u16* __restrict__ wqkvT, u16* __restrict__ woutT) {
  int tid = blockIdx.x * blockDim.x + threadIdx.x;
  int nth = gridDim.x * blockDim.x;
  for (int j = tid; j < 768 * 256; j += nth) {       // wqkvT[n][c] = wqkv[c][n]
    int n = j >> 8, c = j & 255;
    wqkvT[j] = f2bf(wqkv[c * 768 + n]);
  }
  for (int j = tid; j < 256 * 256; j += nth) {       // woutT[n][c] = wout[c][n]
    int n = j >> 8, c = j & 255;
    woutT[j] = f2bf(wout[c * 256 + n]);
  }
}

// ---------------- GroupNorm: one block per (batch, group) ----------------
__global__ void gn_kernel(const float* __restrict__ x, const float* __restrict__ gamma,
                          const float* __restrict__ beta, u16* __restrict__ h) {
  __shared__ float r1[8], r2[8];
  __shared__ float s_mean, s_rstd;
  const int b = blockIdx.x >> 5, g = blockIdx.x & 31;
  const float* xb = x + (size_t)b * SEQ * CH + g * 8;
  float s1 = 0.f, s2 = 0.f;
  for (int i = threadIdx.x; i < SEQ * 8; i += 256) {
    int n = i >> 3, cc = i & 7;
    float v = xb[n * CH + cc];
    s1 += v; s2 += v * v;
  }
  for (int m = 16; m >= 1; m >>= 1) { s1 += __shfl_xor(s1, m, 32); s2 += __shfl_xor(s2, m, 32); }
  int wid = threadIdx.x >> 5, lane = threadIdx.x & 31;
  if (lane == 0) { r1[wid] = s1; r2[wid] = s2; }
  __syncthreads();
  if (threadIdx.x == 0) {
    float t1 = 0.f, t2 = 0.f;
    for (int w = 0; w < 8; w++) { t1 += r1[w]; t2 += r2[w]; }
    float mean = t1 * (1.0f / 32768.0f);
    float var  = t2 * (1.0f / 32768.0f) - mean * mean;
    s_mean = mean; s_rstd = rsqrtf(var + 1e-3f);
  }
  __syncthreads();
  const float mean = s_mean, rstd = s_rstd;
  u16* hb = h + (size_t)b * SEQ * CH + g * 8;
  for (int i = threadIdx.x; i < SEQ * 8; i += 256) {
    int n = i >> 3, cc = i & 7;
    float v = (xb[n * CH + cc] - mean) * rstd * gamma[g * 8 + cc] + beta[g * 8 + cc];
    hb[n * CH + cc] = f2bf(v);
  }
}

// ---------------- QKV GEMM: h[16384,256] x WqkvT -> q,k row-major; v transposed ----------------
__global__ __launch_bounds__(256) void qkv_kernel(const u16* __restrict__ h, const u16* __restrict__ wT,
                                                  const float* __restrict__ bias,
                                                  u16* __restrict__ q, u16* __restrict__ k,
                                                  u16* __restrict__ vT) {
  const int lane = threadIdx.x & 31;
  const int wid  = blockIdx.x * 8 + (threadIdx.x >> 5);   // 49152 tiles: 1024 x 48
  const int tm = wid / 48, tn = wid % 48;
  const int m0 = tm * 16, n0 = tn * 16;
  v8f acc = {};
  for (int k0 = 0; k0 < CH; k0 += 32) {
    v16bf a  = load_a16(h,  CH, m0, k0, lane);
    v16bf bb = load_b16(wT, CH, n0, k0, lane);
    acc = wmma_bf(a, bb, acc);
  }
  const int half = lane >> 4, c = lane & 15;
  const float bv = bias[n0 + c];
  if (n0 < 256) {                                  // ---- Q
    for (int r = 0; r < 8; r++)
      q[(size_t)(m0 + r + half * 8) * CH + n0 + c] = f2bf(acc[r] + bv);
  } else if (n0 < 512) {                           // ---- V -> vT[b][c][n]
    const int b = m0 >> 12, nl = m0 & 4095;
    u32 pk0 = (u32)f2bf(acc[0] + bv) | ((u32)f2bf(acc[1] + bv) << 16);
    u32 pk1 = (u32)f2bf(acc[2] + bv) | ((u32)f2bf(acc[3] + bv) << 16);
    u32 pk2 = (u32)f2bf(acc[4] + bv) | ((u32)f2bf(acc[5] + bv) << 16);
    u32 pk3 = (u32)f2bf(acc[6] + bv) | ((u32)f2bf(acc[7] + bv) << 16);
    uint4 val; val.x = pk0; val.y = pk1; val.z = pk2; val.w = pk3;
    u16* dst = vT + ((size_t)b * CH + (n0 - 256) + c) * SEQ + nl + half * 8;
    *reinterpret_cast<uint4*>(dst) = val;
  } else {                                         // ---- K
    for (int r = 0; r < 8; r++)
      k[(size_t)(m0 + r + half * 8) * CH + n0 - 512 + c] = f2bf(acc[r] + bv);
  }
}

// ---------------- async double-buffered K/V chunk staging ----------------
// K chunk: 32 tokens x 256 ch (row stride KPAD); V chunk: 256 ch x 32 tokens (row stride VPAD)
__device__ __forceinline__ void stage_chunk(const u16* kb, const u16* vb, int mc,
                                            u16* ksb, u16* vsb, int tid) {
  for (int j = 0; j < 8; j++) {                    // 1024 x 16B = K chunk
    int i = tid + j * 128;
    int row = i >> 5, seg = i & 31;
    async_ld16(kb + (size_t)(mc + row) * CH + seg * 8, ksb + row * KPAD + seg * 8);
  }
  for (int j = 0; j < 8; j++) {                    // 1024 x 16B = V chunk
    int i = tid + j * 128;
    int chn = i >> 2, seg = i & 3;
    async_ld16(vb + (size_t)chn * SEQ + mc + seg * 8, vsb + chn * VPAD + seg * 8);
  }
}

// ---------------- Flash attention: 4 waves/block, 16 queries/wave, keys in chunks of 32 ----------------
__global__ __launch_bounds__(128) void attn_kernel(const u16* __restrict__ q, const u16* __restrict__ k,
                                                   const u16* __restrict__ vT, u16* __restrict__ o,
                                                   const float* __restrict__ scale_p) {
  __shared__ __align__(16) u16 ks[2][32 * KPAD];          // 33.8 KB
  __shared__ __align__(16) u16 vs[2][256 * VPAD];         // 41.0 KB
  __shared__ __align__(16) u16 pbuf[4][16 * 40];          // per-wave P tile
  const int tid   = threadIdx.x;
  const int wslot = tid >> 5;
  const int lane  = tid & 31;
  const int half  = lane >> 4, lc = lane & 15;
  const int wt = blockIdx.x * 4 + wslot;                  // 0..1023 query tiles
  const int b  = blockIdx.x >> 6;                         // 64 blocks per batch
  const int q0 = wt * 16;                                 // global token of first query
  const float scale = scale_p[0];

  const u16* kb = k  + (size_t)b * SEQ * CH;
  const u16* vb = vT + (size_t)b * CH * SEQ;

  v16bf qa[8];
  for (int kk = 0; kk < 8; kk++) qa[kk] = load_a16(q, CH, q0, kk * 32, lane);

  v8f zero = {};
  v8f o_acc[16];
  for (int t = 0; t < 16; t++) o_acc[t] = zero;
  float m_i[8], l_i[8];
  for (int r = 0; r < 8; r++) { m_i[r] = -1e30f; l_i[r] = 0.f; }

  u16* pl = &pbuf[wslot][0];

  stage_chunk(kb, vb, 0, &ks[0][0], &vs[0][0], tid);

  for (int ci = 0; ci < SEQ / 32; ci++) {
    const int buf = ci & 1;
    if (ci + 1 < SEQ / 32) {
      stage_chunk(kb, vb, (ci + 1) * 32, &ks[buf ^ 1][0], &vs[buf ^ 1][0], tid);
      WAIT_ASYNC(16);                                     // current chunk's 16 ops done
    } else {
      WAIT_ASYNC(0);
    }
    __syncthreads();                                      // staged data visible to all waves
    const u16* kc = &ks[buf][0];
    const u16* vc = &vs[buf][0];

    // ---- scores: S[16 x 32] = Q . K^T  (K from LDS)
    v8f s0 = zero, s1 = zero;
    for (int kk = 0; kk < 8; kk++) {
      v16bf b0 = load_b16(kc, KPAD, 0,  kk * 32, lane);
      v16bf b1 = load_b16(kc, KPAD, 16, kk * 32, lane);
      s0 = wmma_bf(qa[kk], b0, s0);
      s1 = wmma_bf(qa[kk], b1, s1);
    }
    // ---- online softmax
    float cf[8];
    for (int r = 0; r < 8; r++) {
      float a0 = s0[r] * scale, a1 = s1[r] * scale;
      s0[r] = a0; s1[r] = a1;
      float mx = fmaxf(a0, a1);
      for (int d = 8; d >= 1; d >>= 1) mx = fmaxf(mx, __shfl_xor(mx, d, 32));
      float mnew = fmaxf(m_i[r], mx);
      cf[r] = __expf(m_i[r] - mnew);
      m_i[r] = mnew;
    }
    for (int r = 0; r < 8; r++) {
      float p0 = __expf(s0[r] - m_i[r]);
      float p1 = __expf(s1[r] - m_i[r]);
      s0[r] = p0; s1[r] = p1;
      float sm = p0 + p1;
      for (int d = 8; d >= 1; d >>= 1) sm += __shfl_xor(sm, d, 32);
      l_i[r] = l_i[r] * cf[r] + sm;
    }
    // ---- P: C-layout -> A-layout via per-wave LDS tile (bf16)
    for (int r = 0; r < 8; r++) {
      int row = r + half * 8;
      pl[row * 40 + lc]      = f2bf(s0[r]);
      pl[row * 40 + 16 + lc] = f2bf(s1[r]);
    }
    asm volatile("s_wait_dscnt 0x0" ::: "memory");
    ABOp pa;
    {
      const u16* pp = pl + lc * 40 + half * 8;
      pa.u[0] = *reinterpret_cast<const uint4*>(pp);
      pa.u[1] = *reinterpret_cast<const uint4*>(pp + 16);
    }
    asm volatile("" ::: "memory");
    // ---- rescale O only when some row max moved (EXEC partial only in VALU region)
    float cfs = cf[0] + cf[1] + cf[2] + cf[3] + cf[4] + cf[5] + cf[6] + cf[7];
    if (cfs < 8.0f) {
      for (int t = 0; t < 16; t++)
        for (int r = 0; r < 8; r++) o_acc[t][r] *= cf[r];
    }
    // ---- O += P . V  (V from LDS, key-contiguous B operand)
    for (int t = 0; t < 16; t++) {
      v16bf bv = load_b16(vc, VPAD, t * 16, 0, lane);
      o_acc[t] = wmma_bf(pa.v, bv, o_acc[t]);
    }
    __syncthreads();                                      // all waves done reading this buffer
  }
  // ---- finalize O / l, store bf16 row-major for out-proj
  float inv[8];
  for (int r = 0; r < 8; r++) inv[r] = 1.0f / l_i[r];
  for (int t = 0; t < 16; t++)
    for (int r = 0; r < 8; r++) {
      int tok = q0 + r + half * 8;
      o[(size_t)tok * CH + t * 16 + lc] = f2bf(o_acc[t][r] * inv[r]);
    }
}

// ---------------- out-proj + bias + residual ----------------
__global__ __launch_bounds__(256) void outproj_kernel(const u16* __restrict__ o, const u16* __restrict__ wT,
                                                      const float* __restrict__ bias,
                                                      const float* __restrict__ x, float* __restrict__ out) {
  const int lane = threadIdx.x & 31;
  const int wid  = blockIdx.x * 8 + (threadIdx.x >> 5);   // 16384 tiles: 1024 x 16
  const int tm = wid >> 4, tn = wid & 15;
  const int m0 = tm * 16, n0 = tn * 16;
  v8f acc = {};
  for (int k0 = 0; k0 < CH; k0 += 32) {
    v16bf a  = load_a16(o,  CH, m0, k0, lane);
    v16bf bb = load_b16(wT, CH, n0, k0, lane);
    acc = wmma_bf(a, bb, acc);
  }
  const int half = lane >> 4, c = lane & 15;
  const float bv = bias[n0 + c];
  for (int r = 0; r < 8; r++) {
    size_t idx = (size_t)(m0 + r + half * 8) * CH + n0 + c;
    out[idx] = acc[r] + bv + x[idx];
  }
}

extern "C" void kernel_launch(void* const* d_in, const int* in_sizes, int n_in,
                              void* d_out, int out_size, void* d_ws, size_t ws_size,
                              hipStream_t stream) {
  const float* x     = (const float*)d_in[0];
  const float* gamma = (const float*)d_in[1];
  const float* beta  = (const float*)d_in[2];
  const float* wqkv  = (const float*)d_in[3];
  const float* bqkv  = (const float*)d_in[4];
  const float* scale = (const float*)d_in[5];
  const float* wout  = (const float*)d_in[6];
  const float* bout  = (const float*)d_in[7];
  float* out = (float*)d_out;

  char* ws = (char*)d_ws;
  const size_t TENSOR = (size_t)NTOK * CH * sizeof(u16);  // 8 MB
  u16* h_bf   = (u16*)(ws);                 // reused as O after QKV consumes it
  u16* q_bf   = (u16*)(ws + TENSOR);
  u16* k_bf   = (u16*)(ws + 2 * TENSOR);
  u16* vT_bf  = (u16*)(ws + 3 * TENSOR);
  u16* wqkvT  = (u16*)(ws + 4 * TENSOR);
  u16* woutT  = (u16*)(ws + 4 * TENSOR + 768 * 256 * sizeof(u16));
  u16* o_bf   = h_bf;

  wprep_kernel<<<256, 256, 0, stream>>>(wqkv, wout, wqkvT, woutT);
  gn_kernel<<<BATCH * 32, 256, 0, stream>>>(x, gamma, beta, h_bf);
  qkv_kernel<<<6144, 256, 0, stream>>>(h_bf, wqkvT, bqkv, q_bf, k_bf, vT_bf);
  attn_kernel<<<256, 128, 0, stream>>>(q_bf, k_bf, vT_bf, o_bf, scale);
  outproj_kernel<<<2048, 256, 0, stream>>>(o_bf, woutT, bout, x, out);
}
